// Attention_76192719831597
// MI455X (gfx1250) — matrix operand order repaired and checked by reading, not compiled, not run on hardware
//
#include <hip/hip_runtime.h>

// Problem constants (match reference)
#define B_   4
#define S_   2048
#define H_   2048
#define NH_  16
#define NKV_ 4
#define HD_  128
#define G_   4                      // NH / NKV
#define F_   (H_ + 2 * NKV_ * HD_)  // 3072

typedef __attribute__((ext_vector_type(16))) __bf16       v16bf;
typedef __attribute__((ext_vector_type(8)))  float        v8f;
typedef __attribute__((ext_vector_type(4)))  unsigned int v4u;

union BF16x16 { v16bf bf; v4u q[2]; unsigned short u[16]; };

__device__ __forceinline__ unsigned short f2bfu(float f) {
  unsigned u = __builtin_bit_cast(unsigned, f);
  u += 0x7fffu + ((u >> 16) & 1u);        // round-to-nearest-even
  return (unsigned short)(u >> 16);
}

__device__ __forceinline__ v8f wmma_bf16(v16bf a, v16bf b, v8f c) {
  // D(f32 16x16) = A(bf16 16x32) * B(bf16 32x16) + C
  return __builtin_amdgcn_wmma_f32_16x16x32_bf16(false, a, false, b,
                                                 (short)0, c, false, false);
}

// CDNA5 async global->LDS copy (ASYNCcnt-tracked), 16 bytes per lane.
// LDS byte address = low 32 bits of the generic pointer (aperture mapping).
__device__ __forceinline__ void async_b128(void* lds, const void* g) {
  unsigned l = (unsigned)(unsigned long long)lds;
  asm volatile("global_load_async_to_lds_b128 %0, %1, off"
               :: "v"(l), "v"((unsigned long long)g) : "memory");
}
#define ASYNC_WAIT(imm) asm volatile("s_wait_asynccnt " imm ::: "memory")

// ---------------------------------------------------------------- converts
__global__ __launch_bounds__(256) void cvt_f32_bf16(
    const float* __restrict__ src, unsigned short* __restrict__ dst, int n) {
  int i = blockIdx.x * blockDim.x + threadIdx.x;
  int stride = gridDim.x * blockDim.x;
  for (; i < n; i += stride) dst[i] = f2bfu(src[i]);
}

// src[R][N] f32 -> dst[N][R] bf16 (weights: transpose once so GEMM B tiles
// are contiguous-K and async-stageable)
__global__ __launch_bounds__(256) void cvt_transpose_bf16(
    const float* __restrict__ src, unsigned short* __restrict__ dst,
    int R, int N) {
  int i = blockIdx.x * blockDim.x + threadIdx.x;
  int total = R * N;
  int stride = gridDim.x * blockDim.x;
  for (; i < total; i += stride) {
    int n = i / R, r = i - n * R;
    dst[i] = f2bfu(src[(size_t)r * N + n]);
  }
}

// ---------------------------------------------------------------- GEMM
// C[M,N] f32 = A[M,K] bf16 * Bt[N,K] bf16 (B pre-transposed).
// 256 threads = 8 waves; block tile 128(M) x 128(N); K-step 32.
// Wave tile 32x64: 2 A-frags x 4 B-frags -> 8 WMMA per K-step per wave.
// Double-buffered LDS filled with global_load_async_to_lds_b128.
__global__ __launch_bounds__(256) void gemm_bf16(
    const unsigned short* __restrict__ A, const unsigned short* __restrict__ Bt,
    float* __restrict__ C, int M, int N, int K) {
  __shared__ alignas(16) unsigned short As[2][128 * 40];  // [row][k]
  __shared__ alignas(16) unsigned short Bs[2][128 * 40];  // [col][k]

  const int tid  = threadIdx.x;
  const int lane = tid & 31;
  const int w    = tid >> 5;
  const int wave_m = w >> 1, wave_n = w & 1;
  const int m0 = blockIdx.y * 128;
  const int n0 = blockIdx.x * 128;
  const int kl = lane & 15;

  v8f acc[2][4];
#pragma unroll
  for (int mi = 0; mi < 2; ++mi)
#pragma unroll
    for (int n = 0; n < 4; ++n)
#pragma unroll
      for (int i = 0; i < 8; ++i) acc[mi][n][i] = 0.f;

  auto stage = [&](int k0, int buf) {
    // A and B tiles each 128x32 bf16 = 512 x 16B chunks -> 2+2 per thread
#pragma unroll
    for (int it = 0; it < 2; ++it) {
      int cid = tid + it * 256;
      int row = cid >> 2, c8 = (cid & 3) * 8;
      async_b128(&As[buf][row * 40 + c8],
                 A + (size_t)(m0 + row) * K + k0 + c8);
      async_b128(&Bs[buf][row * 40 + c8],
                 Bt + (size_t)(n0 + row) * K + k0 + c8);
    }
  };

  stage(0, 0);
  for (int k0 = 0; k0 < K; k0 += 32) {
    const int c = (k0 >> 5) & 1;
    if (k0 + 32 < K) { stage(k0 + 32, 1 - c); ASYNC_WAIT("0x4"); }
    else             { ASYNC_WAIT("0x0"); }
    __syncthreads();

    const unsigned short* Asc = As[c];
    const unsigned short* Bsc = Bs[c];
    const int khalf = (lane >> 4) * 8;
    const int koff  = (lane >> 4) * 8;
    // 2 A fragments (16x32 bf16, ISA layout)
    BF16x16 af[2];
#pragma unroll
    for (int mi = 0; mi < 2; ++mi) {
      const int arow = 32 * wave_m + 16 * mi + kl;
      af[mi].q[0] = *(const v4u*)&Asc[arow * 40 + khalf];
      af[mi].q[1] = *(const v4u*)&Asc[arow * 40 + 16 + khalf];
    }
    // 4 B fragments (lane = column, elems = K in ISA table order)
    BF16x16 bfr[4];
#pragma unroll
    for (int n = 0; n < 4; ++n) {
      const int col = 64 * wave_n + 16 * n + kl;
      bfr[n].q[0] = *(const v4u*)&Bsc[col * 40 + koff];
      bfr[n].q[1] = *(const v4u*)&Bsc[col * 40 + koff + 16];
    }
    // 8 WMMA with full fragment reuse
#pragma unroll
    for (int mi = 0; mi < 2; ++mi)
#pragma unroll
      for (int n = 0; n < 4; ++n)
        acc[mi][n] = wmma_bf16(af[mi].bf, bfr[n].bf, acc[mi][n]);
    __syncthreads();
  }

#pragma unroll
  for (int mi = 0; mi < 2; ++mi)
#pragma unroll
    for (int n = 0; n < 4; ++n)
#pragma unroll
      for (int i = 0; i < 8; ++i) {
        int row = m0 + 32 * wave_m + 16 * mi + i + ((lane >> 4) << 3);
        int col = n0 + 64 * wave_n + 16 * n + kl;
        C[(size_t)row * N + col] = acc[mi][n][i];
      }
}

// ---------------------------------------------------------------- RoPE
// qkv[B,S,F] f32 -> Q[B,NH,S,HD], K[B,NKV,S,HD] bf16 (row-major) and
// V transposed [B,NKV,HD,S] bf16 so flash V tiles are contiguous copies.
__global__ __launch_bounds__(256) void rope_scatter(
    const float* __restrict__ qkv, const float* __restrict__ cosT,
    const float* __restrict__ sinT, unsigned short* __restrict__ qbuf,
    unsigned short* __restrict__ kbuf, unsigned short* __restrict__ vbt) {
  int idx = blockIdx.x * blockDim.x + threadIdx.x;
  int d  = idx & 63;
  int j  = (idx >> 6) % 6;                     // 0..3 q heads, 4 = k, 5 = v
  int kv = (idx >> 6) / 6 % NKV_;
  int s  = (idx >> 6) / (6 * NKV_) % S_;
  int b  = (idx >> 6) / (6 * NKV_ * S_);

  size_t row = (size_t)(b * S_ + s) * F_ + kv * ((G_ + 2) * HD_) + j * HD_;
  float x1 = qkv[row + d];
  float x2 = qkv[row + d + 64];
  float o1 = x1, o2 = x2;
  if (j < 5) {  // rotate q and k
    float c1 = cosT[s * HD_ + d],      s1 = sinT[s * HD_ + d];
    float c2 = cosT[s * HD_ + d + 64], s2 = sinT[s * HD_ + d + 64];
    o1 = x1 * c1 - x2 * s1;
    o2 = x2 * c2 + x1 * s2;
  }
  if (j < G_) {
    int h = kv * G_ + j;
    size_t o = ((size_t)(b * NH_ + h) * S_ + s) * HD_;
    qbuf[o + d] = f2bfu(o1); qbuf[o + d + 64] = f2bfu(o2);
  } else if (j == G_) {
    size_t o = ((size_t)(b * NKV_ + kv) * S_ + s) * HD_;
    kbuf[o + d] = f2bfu(o1); kbuf[o + d + 64] = f2bfu(o2);
  } else {
    size_t o = ((size_t)(b * NKV_ + kv) * HD_ + d) * S_ + s;
    vbt[o] = f2bfu(o1); vbt[o + (size_t)64 * S_] = f2bfu(o2);
  }
}

// ---------------------------------------------------------------- flash attn
// grid (S/64, B*NH); 4 waves/block; wave owns 16 query rows x HD=128.
// K/V tiles double-buffered in LDS via async copies.
__global__ __launch_bounds__(128) void flash_attn(
    const unsigned short* __restrict__ qbuf, const unsigned short* __restrict__ kbuf,
    const unsigned short* __restrict__ vbt, unsigned short* __restrict__ aobuf) {
  __shared__ alignas(16) unsigned short Ks[2][32 * 136];   // [key][d]
  __shared__ alignas(16) unsigned short Vs[2][128 * 40];   // [d][key]
  __shared__ alignas(16) unsigned short Ps[4 * 16 * 40];   // per-wave P scratch

  const int tid  = threadIdx.x;
  const int lane = tid & 31;
  const int w    = tid >> 5;
  const int bh = blockIdx.y;
  const int b = bh / NH_, h = bh % NH_, kv = h / G_;
  const int q0 = blockIdx.x * 64;
  const int qb = q0 + w * 16;
  unsigned short* Pw = &Ps[w * (16 * 40)];
  const int kl   = lane & 15;
  const int koff = (lane >> 4) * 8;
  const float scale = 0.08838834764831845f;  // 1/sqrt(128)

  // persistent Q fragments: 4 x (16x32 bf16) covering HD=128
  BF16x16 qf[4];
  {
    const unsigned short* qp =
        qbuf + ((size_t)(b * NH_ + h) * S_ + (qb + kl)) * HD_;
    const int khalf = (lane >> 4) * 8;
#pragma unroll
    for (int kk = 0; kk < 4; ++kk) {
      qf[kk].q[0] = *(const v4u*)(qp + kk * 32 + khalf);
      qf[kk].q[1] = *(const v4u*)(qp + kk * 32 + 16 + khalf);
    }
  }

  const size_t kbase = (size_t)(b * NKV_ + kv) * S_ * HD_;          // kbuf [s][d]
  const size_t vbase = (size_t)(b * NKV_ + kv) * HD_ * (size_t)S_;  // vbt [d][s]
  auto stage = [&](int key0, int buf) {
#pragma unroll
    for (int it = 0; it < 4; ++it) {
      int cid = tid + it * 128;                 // 0..511
      int r  = cid >> 4, c8 = (cid & 15) * 8;   // K tile chunk
      async_b128(&Ks[buf][r * 136 + c8],
                 kbuf + kbase + (size_t)(key0 + r) * HD_ + c8);
      int dd = cid >> 2, k8 = (cid & 3) * 8;    // V tile chunk
      async_b128(&Vs[buf][dd * 40 + k8],
                 vbt + vbase + (size_t)dd * S_ + key0 + k8);
    }
  };

  v8f acc[8];
  float m[8], l[8];
#pragma unroll
  for (int i = 0; i < 8; ++i) { m[i] = -1e30f; l[i] = 0.f; }
#pragma unroll
  for (int n = 0; n < 8; ++n)
#pragma unroll
    for (int i = 0; i < 8; ++i) acc[n][i] = 0.f;

  const int nkb = q0 / 32 + 2;  // causal: key blocks covering rows <= q0+63
  stage(0, 0);
  for (int kb = 0; kb < nkb; ++kb) {
    const int c = kb & 1;
    const int key0 = kb * 32;
    if (kb + 1 < nkb) { stage(key0 + 32, 1 - c); ASYNC_WAIT("0x8"); }
    else              { ASYNC_WAIT("0x0"); }
    __syncthreads();
    const unsigned short* Kc = Ks[c];
    const unsigned short* Vc = Vs[c];

    // scores: 16 queries x 32 keys = two C fragments, 8 WMMA over d=128
    v8f s0, s1;
#pragma unroll
    for (int i = 0; i < 8; ++i) { s0[i] = 0.f; s1[i] = 0.f; }
#pragma unroll
    for (int kk = 0; kk < 4; ++kk) {
      BF16x16 b0, b1;  // B = K^T: lane = key column, elems = d in ISA table order
      b0.q[0] = *(const v4u*)&Kc[kl * 136 + kk * 32 + koff];
      b0.q[1] = *(const v4u*)&Kc[kl * 136 + kk * 32 + koff + 16];
      b1.q[0] = *(const v4u*)&Kc[(16 + kl) * 136 + kk * 32 + koff];
      b1.q[1] = *(const v4u*)&Kc[(16 + kl) * 136 + kk * 32 + koff + 16];
      s0 = wmma_bf16(qf[kk].bf, b0.bf, s0);
      s1 = wmma_bf16(qf[kk].bf, b1.bf, s1);
    }

    // online softmax (half-wave xor reductions; wave32)
#pragma unroll
    for (int i = 0; i < 8; ++i) {
      int row = qb + i + ((lane >> 4) << 3);
      int c0 = key0 + kl, c1 = c0 + 16;
      float a0 = s0[i] * scale + ((c0 > row) ? -1e30f : 0.f);
      float a1 = s1[i] * scale + ((c1 > row) ? -1e30f : 0.f);
      float mx = fmaxf(a0, a1);
#pragma unroll
      for (int off = 8; off >= 1; off >>= 1)
        mx = fmaxf(mx, __shfl_xor(mx, off, 32));
      float mnew = fmaxf(m[i], mx);
      float corr = __expf(m[i] - mnew);
      float p0 = __expf(a0 - mnew);
      float p1 = __expf(a1 - mnew);
      float rs = p0 + p1;
#pragma unroll
      for (int off = 8; off >= 1; off >>= 1)
        rs += __shfl_xor(rs, off, 32);
      l[i] = l[i] * corr + rs;
      m[i] = mnew;
#pragma unroll
      for (int n = 0; n < 8; ++n) acc[n][i] *= corr;
      s0[i] = p0; s1[i] = p1;
    }

    // C-layout P -> A-fragment layout via wave-private LDS transpose
#pragma unroll
    for (int i = 0; i < 8; ++i) {
      int r = i + ((lane >> 4) << 3);
      Pw[r * 40 + kl]      = f2bfu(s0[i]);
      Pw[r * 40 + 16 + kl] = f2bfu(s1[i]);
    }
    BF16x16 pa;
    {
      const int khalf = (lane >> 4) * 8;
      pa.q[0] = *(const v4u*)&Pw[kl * 40 + khalf];
      pa.q[1] = *(const v4u*)&Pw[kl * 40 + 16 + khalf];
    }
    // P(16x32) x V(32x128): 8 WMMA reusing the same A fragment
#pragma unroll
    for (int n = 0; n < 8; ++n) {
      BF16x16 bv;
      int dcol = n * 16 + kl;
      bv.q[0] = *(const v4u*)&Vc[dcol * 40 + koff];
      bv.q[1] = *(const v4u*)&Vc[dcol * 40 + koff + 16];
      acc[n] = wmma_bf16(pa.bf, bv.bf, acc[n]);
    }
    __syncthreads();
  }

  // write attn output bf16 [B,S,NH*HD] (A matrix of the out-projection)
#pragma unroll
  for (int i = 0; i < 8; ++i) {
    int srow = qb + i + ((lane >> 4) << 3);
    float invl = 1.f / l[i];
    size_t o = (size_t)(b * S_ + srow) * (NH_ * HD_) + h * HD_;
#pragma unroll
    for (int n = 0; n < 8; ++n)
      aobuf[o + n * 16 + kl] = f2bfu(acc[n][i] * invl);
  }
}

// ---------------------------------------------------------------- launch
extern "C" void kernel_launch(void* const* d_in, const int* in_sizes, int n_in,
                              void* d_out, int out_size, void* d_ws, size_t ws_size,
                              hipStream_t stream) {
  (void)in_sizes; (void)n_in; (void)out_size; (void)ws_size;
  const float* hidden   = (const float*)d_in[0];
  // d_in[1] = attention_mask: causal mask computed analytically, never read
  const float* rope_cos = (const float*)d_in[2];
  const float* rope_sin = (const float*)d_in[3];
  const float* w_attn   = (const float*)d_in[4];
  const float* w_proj   = (const float*)d_in[5];

  char* ws = (char*)d_ws;
  size_t off = 0;
  auto wsalloc = [&](size_t bytes) -> void* {
    void* p = ws + off;
    off += (bytes + 255) & ~(size_t)255;
    return p;
  };
  unsigned short* wattn_t = (unsigned short*)wsalloc((size_t)H_ * F_ * 2);  // [F][H]
  unsigned short* wproj_t = (unsigned short*)wsalloc((size_t)H_ * H_ * 2);  // [H][H]
  unsigned short* hs_bf   = (unsigned short*)wsalloc((size_t)B_ * S_ * H_ * 2);
  float*          qkv     = (float*)wsalloc((size_t)B_ * S_ * F_ * 4);
  unsigned short* qbuf    = (unsigned short*)wsalloc((size_t)B_ * NH_ * S_ * HD_ * 2);
  unsigned short* kbuf    = (unsigned short*)wsalloc((size_t)B_ * NKV_ * S_ * HD_ * 2);
  unsigned short* vbt     = (unsigned short*)wsalloc((size_t)B_ * NKV_ * S_ * HD_ * 2);
  unsigned short* aobuf   = (unsigned short*)wsalloc((size_t)B_ * S_ * H_ * 2);

  cvt_transpose_bf16<<<4096, 256, 0, stream>>>(w_attn, wattn_t, H_, F_);
  cvt_transpose_bf16<<<4096, 256, 0, stream>>>(w_proj, wproj_t, H_, H_);
  cvt_f32_bf16<<<4096, 256, 0, stream>>>(hidden, hs_bf, B_ * S_ * H_);

  // qkv = hidden x w_attn : M=8192, N=3072, K=2048
  gemm_bf16<<<dim3(F_ / 128, (B_ * S_) / 128), 256, 0, stream>>>(
      hs_bf, wattn_t, qkv, B_ * S_, F_, H_);

  rope_scatter<<<(B_ * S_ * NKV_ * 6 * 64) / 256, 256, 0, stream>>>(
      qkv, rope_cos, rope_sin, qbuf, kbuf, vbt);

  flash_attn<<<dim3(S_ / 64, B_ * NH_), 128, 0, stream>>>(
      qbuf, kbuf, vbt, aobuf);

  // out = attn_out x w_proj : M=8192, N=2048, K=2048
  gemm_bf16<<<dim3(H_ / 128, (B_ * S_) / 128), 256, 0, stream>>>(
      aobuf, wproj_t, (float*)d_out, B_ * S_, H_, H_);
}
